// AttentionV2_65180423685285
// MI455X (gfx1250) — matrix operand order repaired and checked by reading, not compiled, not run on hardware
//
#include <hip/hip_runtime.h>

#define BATCH 4
#define CDIM  256
#define HEADS 8
#define HD    32
#define HW    48
#define NPIX  2304              // 48*48
#define NG    (BATCH * NPIX)    // 9216
#define SCALE 0.17677669529663687f  // 32^-0.5

typedef __attribute__((ext_vector_type(16))) __bf16 v16bf;
typedef __attribute__((ext_vector_type(8)))  __bf16 v8bf;
typedef __attribute__((ext_vector_type(8)))  float  v8f;

union ABu { v16bf v; v8bf h[2]; };

static __device__ __forceinline__ v8f wmma_bf16f32(v16bf a, v16bf b, v8f c) {
    // D = A(16x32 bf16) * B(32x16 bf16) + C(16x16 f32)
    return __builtin_amdgcn_wmma_f32_16x16x32_bf16(
        /*neg_a=*/false, a, /*neg_b=*/false, b,
        /*c_mod=*/(short)0, c, /*reuse_a=*/false, /*reuse_b=*/false);
}

// ---------------- prep kernels ----------------

__global__ void cvt_f32_bf16(const float* __restrict__ src, __bf16* __restrict__ dst, int n) {
    int i = blockIdx.x * blockDim.x + threadIdx.x;
    if (i < n) dst[i] = (__bf16)src[i];
}

// x[b][c][n] f32 -> xt[(b*NPIX+n)][c] bf16
__global__ void transpose_x(const float* __restrict__ x, __bf16* __restrict__ xt) {
    int idx = blockIdx.x * blockDim.x + threadIdx.x;   // over NG*CDIM
    int c  = idx & (CDIM - 1);
    int gn = idx >> 8;
    int b  = gn / NPIX;
    int n  = gn % NPIX;
    xt[idx] = (__bf16)x[((size_t)b * CDIM + c) * NPIX + n];
}

// ---------------- fused QKV projection GEMM ----------------
// M = 768 (512 qk + 256 v), K = 256, N = 9216.
// One wave: 16(M) x 64(N) tile; A operand reused across 4 WMMAs per K-step.
__global__ __launch_bounds__(128) void qkv_gemm(
    const __bf16* __restrict__ xt, const __bf16* __restrict__ wqk,
    const __bf16* __restrict__ wv, const float* __restrict__ b_qk,
    const float* __restrict__ b_v, __bf16* __restrict__ qb,
    __bf16* __restrict__ kbuf, __bf16* __restrict__ vt,
    float* __restrict__ v4)
{
    const int lane = threadIdx.x & 31;
    const int wave = threadIdx.x >> 5;
    const int l16  = lane & 15;
    const int half = lane >> 4;
    const int job  = blockIdx.x * 4 + wave;     // 48 Mtiles * 144 N64-tiles = 6912
    const int mt   = job % 48;
    const int n0   = (job / 48) * 64;

    const __bf16* W;
    int m0;
    if (mt < 32) { W = wqk; m0 = mt * 16; }
    else         { W = wv;  m0 = (mt - 32) * 16; }

    v8f acc[4] = {};
    for (int k0 = 0; k0 < CDIM; k0 += 32) {
        // A operand: row m0+l16, K chunk [k0 + half*16, +16)
        const __bf16* pa = W + (size_t)(m0 + l16) * CDIM + k0 + half * 16;
        ABu ua; ua.h[0] = *(const v8bf*)pa; ua.h[1] = *(const v8bf*)(pa + 8);
#pragma unroll
        for (int t = 0; t < 4; ++t) {
            const __bf16* pb = xt + (size_t)(n0 + t * 16 + l16) * CDIM + k0 + half * 16;
            v16bf vb = *(const v16bf*)pb;
            acc[t] = wmma_bf16f32(ua.v, vb, acc[t]);
        }
    }

#pragma unroll
    for (int t = 0; t < 4; ++t) {
#pragma unroll
        for (int r = 0; r < 8; ++r) {
            int o  = m0 + r + half * 8;
            int gn = n0 + t * 16 + l16;
            int b  = gn / NPIX;
            int n  = gn % NPIX;
            float val = acc[t][r];
            if (mt < 32) {
                val += b_qk[o];
                int head = o >> 6, w = o & 63;
                size_t bh = (size_t)b * HEADS + head;
                if (w < HD) qb  [(bh * NPIX + n) * HD + w]        = (__bf16)val;
                else        kbuf[(bh * NPIX + n) * HD + (w - HD)] = (__bf16)val;
            } else {
                val += b_v[o];
                v4[((size_t)b * CDIM + o) * NPIX + n] = val;
                int head = o >> 5, dd = o & 31;
                vt[(((size_t)b * HEADS + head) * HD + dd) * NPIX + n] = (__bf16)val;
            }
        }
    }
}

// ---------------- depthwise 3x3 conv (positional encoding) ----------------
__global__ void dwconv3x3(const float* __restrict__ v4, const float* __restrict__ w,
                          const float* __restrict__ bias, float* __restrict__ pe)
{
    int idx = blockIdx.x * blockDim.x + threadIdx.x;   // over BATCH*CDIM*NPIX
    int x = idx % HW;
    int y = (idx / HW) % HW;
    int c = (idx / NPIX) % CDIM;
    int b = idx / (NPIX * CDIM);
    const float* wp  = w + c * 9;
    const float* src = v4 + ((size_t)b * CDIM + c) * NPIX;
    float s = bias[c];
#pragma unroll
    for (int dy = -1; dy <= 1; ++dy) {
        int yy = y + dy;
        if (yy < 0 || yy >= HW) continue;
#pragma unroll
        for (int dx = -1; dx <= 1; ++dx) {
            int xx = x + dx;
            if (xx < 0 || xx >= HW) continue;
            s += wp[(dy + 1) * 3 + (dx + 1)] * src[yy * HW + xx];
        }
    }
    pe[idx] = s;
}

// ---------------- flash attention ----------------
// One wave (32 threads) per (b,h, 16-query-row tile). d = 32 -> one K-step per WMMA.
__global__ __launch_bounds__(32) void flash_attn(
    const __bf16* __restrict__ qb, const __bf16* __restrict__ kbuf,
    const __bf16* __restrict__ vt, float* __restrict__ outp)
{
    const int lane = threadIdx.x & 31;
    const int l16  = lane & 15;
    const int half = lane >> 4;
    const int bh   = blockIdx.x / (NPIX / 16);
    const int q0   = (blockIdx.x % (NPIX / 16)) * 16;

    const __bf16* Q = qb   + (size_t)bh * NPIX * HD;
    const __bf16* K = kbuf + (size_t)bh * NPIX * HD;
    const __bf16* V = vt   + (size_t)bh * HD * NPIX;

    __shared__ alignas(32) __bf16 pbuf[16 * 32];

    // Q A-operand (16 rows x 32 K, bf16)
    const __bf16* pq = Q + (size_t)(q0 + l16) * HD + half * 16;
    ABu qa; qa.h[0] = *(const v8bf*)pq; qa.h[1] = *(const v8bf*)(pq + 8);

    v8f O0 = {}, O1 = {};
    float mrun[8], lrun[8];
#pragma unroll
    for (int r = 0; r < 8; ++r) { mrun[r] = -3.0e38f; lrun[r] = 0.f; }

    for (int j = 0; j < NPIX / 32; ++j) {
        const int kb0 = j * 32;
        v16bf B0 = *(const v16bf*)(K + (size_t)(kb0 + l16) * HD + half * 16);
        v16bf B1 = *(const v16bf*)(K + (size_t)(kb0 + 16 + l16) * HD + half * 16);
        v8f z = {};
        v8f S0 = wmma_bf16f32(qa.v, B0, z);
        v8f S1 = wmma_bf16f32(qa.v, B1, z);

#pragma unroll
        for (int r = 0; r < 8; ++r) {
            float s0 = S0[r] * SCALE;
            float s1 = S1[r] * SCALE;
            float mx = fmaxf(s0, s1);
#pragma unroll
            for (int msk = 1; msk <= 8; msk <<= 1)
                mx = fmaxf(mx, __shfl_xor(mx, msk, 32));
            float mnew  = fmaxf(mrun[r], mx);
            float alpha = __expf(mrun[r] - mnew);
            float p0 = __expf(s0 - mnew);
            float p1 = __expf(s1 - mnew);
            float rs = p0 + p1;
#pragma unroll
            for (int msk = 1; msk <= 8; msk <<= 1)
                rs += __shfl_xor(rs, msk, 32);
            lrun[r] = lrun[r] * alpha + rs;
            mrun[r] = mnew;
            O0[r] *= alpha;
            O1[r] *= alpha;
            int m = r + half * 8;                 // C-layout row
            pbuf[m * 32 + l16]      = (__bf16)p0; // keys kb0..kb0+15
            pbuf[m * 32 + 16 + l16] = (__bf16)p1; // keys kb0+16..kb0+31
        }
        __syncthreads();
        // P A-operand from LDS (16 q-rows x 32 keys)
        const __bf16* pp = pbuf + l16 * 32 + half * 16;
        ABu pa; pa.h[0] = *(const v8bf*)pp; pa.h[1] = *(const v8bf*)(pp + 8);
        __syncthreads();
        // V B-operands: Vt[dcol][key] -> contiguous along keys
        v16bf Vb0 = *(const v16bf*)(V + (size_t)l16 * NPIX        + kb0 + half * 16);
        v16bf Vb1 = *(const v16bf*)(V + (size_t)(16 + l16) * NPIX + kb0 + half * 16);
        O0 = wmma_bf16f32(pa.v, Vb0, O0);
        O1 = wmma_bf16f32(pa.v, Vb1, O1);
    }

#pragma unroll
    for (int r = 0; r < 8; ++r) {
        float inv = 1.0f / lrun[r];
        O0[r] *= inv;
        O1[r] *= inv;
    }
    // store: lane holds fixed dcol, n contiguous across r -> float4 stores
    float* o0p = outp + ((size_t)bh * HD + l16)      * NPIX + q0 + half * 8;
    float* o1p = outp + ((size_t)bh * HD + 16 + l16) * NPIX + q0 + half * 8;
    *(float4*)(o0p)     = make_float4(O0[0], O0[1], O0[2], O0[3]);
    *(float4*)(o0p + 4) = make_float4(O0[4], O0[5], O0[6], O0[7]);
    *(float4*)(o1p)     = make_float4(O1[0], O1[1], O1[2], O1[3]);
    *(float4*)(o1p + 4) = make_float4(O1[4], O1[5], O1[6], O1[7]);
}

// ---------------- (attn_out + pe) -> transposed bf16 [gn][c] ----------------
__global__ void add_pe_transpose(const float* __restrict__ attn, const float* __restrict__ pe,
                                 __bf16* __restrict__ t)
{
    int idx = blockIdx.x * blockDim.x + threadIdx.x;   // over NG*CDIM
    int c  = idx & (CDIM - 1);
    int gn = idx >> 8;
    int b  = gn / NPIX;
    int n  = gn % NPIX;
    size_t src = ((size_t)b * CDIM + c) * NPIX + n;
    t[idx] = (__bf16)(attn[src] + pe[src]);
}

// ---------------- final projection GEMM ----------------
// M = 256, K = 256, N = 9216. Wave computes 16x64.
__global__ __launch_bounds__(128) void proj_gemm(
    const __bf16* __restrict__ t, const __bf16* __restrict__ wp,
    const float* __restrict__ bias, float* __restrict__ out)
{
    const int lane = threadIdx.x & 31;
    const int wave = threadIdx.x >> 5;
    const int l16  = lane & 15;
    const int half = lane >> 4;
    const int job  = blockIdx.x * 4 + wave;     // 16 Mtiles * 144 = 2304 jobs
    const int mt   = job % 16;
    const int n0   = (job / 16) * 64;
    const int m0   = mt * 16;

    v8f acc[4] = {};
    for (int k0 = 0; k0 < CDIM; k0 += 32) {
        const __bf16* pa = wp + (size_t)(m0 + l16) * CDIM + k0 + half * 16;
        ABu ua; ua.h[0] = *(const v8bf*)pa; ua.h[1] = *(const v8bf*)(pa + 8);
#pragma unroll
        for (int tt = 0; tt < 4; ++tt) {
            const __bf16* pb = t + (size_t)(n0 + tt * 16 + l16) * CDIM + k0 + half * 16;
            v16bf vb = *(const v16bf*)pb;
            acc[tt] = wmma_bf16f32(ua.v, vb, acc[tt]);
        }
    }
#pragma unroll
    for (int tt = 0; tt < 4; ++tt) {
#pragma unroll
        for (int r = 0; r < 8; ++r) {
            int o  = m0 + r + half * 8;
            int gn = n0 + tt * 16 + l16;
            int b  = gn / NPIX;
            int n  = gn % NPIX;
            out[((size_t)b * CDIM + o) * NPIX + n] = acc[tt][r] + bias[o];
        }
    }
}

// ---------------- launch ----------------
extern "C" void kernel_launch(void* const* d_in, const int* in_sizes, int n_in,
                              void* d_out, int out_size, void* d_ws, size_t ws_size,
                              hipStream_t stream)
{
    const float* x      = (const float*)d_in[0];
    const float* w_qk   = (const float*)d_in[1];
    const float* b_qk   = (const float*)d_in[2];
    const float* w_v    = (const float*)d_in[3];
    const float* b_v    = (const float*)d_in[4];
    const float* w_pe   = (const float*)d_in[5];
    const float* b_pe   = (const float*)d_in[6];
    const float* w_proj = (const float*)d_in[7];
    const float* b_proj = (const float*)d_in[8];
    float* out = (float*)d_out;

    char* ws = (char*)d_ws;
    size_t off = 0;
    auto carve = [&](size_t bytes) { char* p = ws + off; off += (bytes + 255) & ~(size_t)255; return p; };

    __bf16* xt     = (__bf16*)carve((size_t)NG * CDIM * 2);          // 4.5 MB
    __bf16* wqk_bf = (__bf16*)carve((size_t)2 * CDIM * CDIM * 2);    // 256 KB
    __bf16* wv_bf  = (__bf16*)carve((size_t)CDIM * CDIM * 2);        // 128 KB
    __bf16* wp_bf  = (__bf16*)carve((size_t)CDIM * CDIM * 2);        // 128 KB
    __bf16* qb     = (__bf16*)carve((size_t)BATCH * HEADS * NPIX * HD * 2);
    __bf16* kbuf   = (__bf16*)carve((size_t)BATCH * HEADS * NPIX * HD * 2);
    __bf16* vt     = (__bf16*)carve((size_t)BATCH * HEADS * HD * NPIX * 2);
    float*  v4     = (float*) carve((size_t)BATCH * CDIM * NPIX * 4);
    float*  pe     = (float*) carve((size_t)BATCH * CDIM * NPIX * 4);
    float*  attn   = (float*) carve((size_t)BATCH * CDIM * NPIX * 4);
    __bf16* tbuf   = (__bf16*)carve((size_t)NG * CDIM * 2);
    (void)ws_size; (void)in_sizes; (void)n_in; (void)out_size; (void)b_pe;

    // 1. weight conversions + x transpose
    cvt_f32_bf16<<<(2 * CDIM * CDIM + 255) / 256, 256, 0, stream>>>(w_qk, wqk_bf, 2 * CDIM * CDIM);
    cvt_f32_bf16<<<(CDIM * CDIM + 255) / 256, 256, 0, stream>>>(w_v, wv_bf, CDIM * CDIM);
    cvt_f32_bf16<<<(CDIM * CDIM + 255) / 256, 256, 0, stream>>>(w_proj, wp_bf, CDIM * CDIM);
    transpose_x<<<NG * CDIM / 256, 256, 0, stream>>>(x, xt);

    // 2. fused qkv GEMM (48 Mtiles * 144 N64-tiles = 6912 wave jobs / 4 waves per block)
    qkv_gemm<<<6912 / 4, 128, 0, stream>>>(xt, wqk_bf, wv_bf, b_qk, b_v, qb, kbuf, vt, v4);

    // 3. depthwise 3x3 conv
    dwconv3x3<<<BATCH * CDIM * NPIX / 256, 256, 0, stream>>>(v4, w_pe, b_pe, pe);

    // 4. flash attention: one wave per (b,h,16-row tile)
    flash_attn<<<BATCH * HEADS * (NPIX / 16), 32, 0, stream>>>(qb, kbuf, vt, attn);

    // 5. add pe + transpose to bf16 [gn][c]
    add_pe_transpose<<<NG * CDIM / 256, 256, 0, stream>>>(attn, pe, tbuf);

    // 6. final projection GEMM (16 Mtiles * 144 = 2304 wave jobs / 4 per block)
    proj_gemm<<<2304 / 4, 128, 0, stream>>>(tbuf, wp_bf, b_proj, out);
}
